// MultiHeadAttention_49546742726675
// MI455X (gfx1250) — compile-verified, matrix operand order checked by reading, not run on hardware
//
#include <hip/hip_runtime.h>
#include <hip/hip_bf16.h>

typedef __attribute__((ext_vector_type(16))) __bf16 v16bf;
typedef __attribute__((ext_vector_type(8)))  __bf16 v8bf;
typedef __attribute__((ext_vector_type(8)))  float  v8f;
typedef int v4i_b __attribute__((vector_size(16)));  // b128 payload type

static constexpr int HEADS = 8, D_MODEL = 512, DK = 64, SEQ = 2048, BATCH = 4;
static constexpr int MTOK = BATCH * SEQ;  // 8192 tokens

// ---------- gfx1250 async global->LDS path (guarded; falls back cleanly) ----
#if defined(__AMDGCN__) && __has_builtin(__builtin_amdgcn_global_load_async_to_lds_b128)
#define HAVE_ASYNC_LDS 1
__device__ __forceinline__ void async_lds_b128(const void* gsrc, void* ldst) {
  __builtin_amdgcn_global_load_async_to_lds_b128(
      (__attribute__((address_space(1))) v4i_b*)(gsrc),
      (__attribute__((address_space(3))) v4i_b*)(ldst),
      0, 0);
}
__device__ __forceinline__ void wait_async() {
#if __has_builtin(__builtin_amdgcn_s_wait_asynccnt)
  __builtin_amdgcn_s_wait_asynccnt(0);
#else
  asm volatile("s_wait_asynccnt 0x0" ::: "memory");
#endif
}
#else
#define HAVE_ASYNC_LDS 0
#endif

// ---------- helpers ----------

__device__ __forceinline__ __bf16 f2bf(float f) {
  unsigned u = __builtin_bit_cast(unsigned, f);
  unsigned r = (u + 0x7FFFu + ((u >> 16) & 1u)) >> 16;  // RNE
  return __builtin_bit_cast(__bf16, (unsigned short)r);
}

// Load one 16-bit WMMA operand fragment (A 16xK=32 row-major, or B KxN with
// lane = n reading contiguous K). CDNA5 layout: lanes 0-15 hold K 0-7 & 16-23,
// lanes 16-31 hold K 8-15 & 24-31 (2 elems per VGPR, 8 VGPRs).
__device__ __forceinline__ v16bf load_frag(const __bf16* base, int ld) {
  const int lane = threadIdx.x & 31;
  const int half = lane >> 4;
  const __bf16* p = base + (size_t)(lane & 15) * ld + half * 8;
  v8bf lo = *(const v8bf*)(p);
  v8bf hi = *(const v8bf*)(p + 16);
  v16bf o;
#pragma unroll
  for (int i = 0; i < 8; ++i) { o[i] = lo[i]; o[i + 8] = hi[i]; }
  return o;
}

// ---------- kernel 0: f32 -> bf16 cast ----------

__global__ void convert_f32_bf16(const float* __restrict__ in,
                                 __bf16* __restrict__ out, int n) {
  int i = blockIdx.x * blockDim.x + threadIdx.x;
  if (i < n) out[i] = f2bf(in[i]);
}

// ---------- kernels 1 & 3: C[M,N] = A[M,K] @ B[N,K]^T + bias ----------
// Block tile 128x64, BK=32 (one wmma k-step). 8 waves, each owns 32x32.

template <bool OUT_BF16>
__global__ __launch_bounds__(256) void gemm_bias_wmma(
    const __bf16* __restrict__ A, const __bf16* __restrict__ B,
    const float* __restrict__ bias, void* __restrict__ Cout,
    int M, int N, int K) {
  constexpr int BM = 128, BN = 64, BK = 32, LDT = 48;  // 96B rows: 16B aligned
  __shared__ __align__(16) __bf16 As[BM][LDT];
  __shared__ __align__(16) __bf16 Bs[BN][LDT];
  const int t = threadIdx.x, lane = t & 31, w = t >> 5;
  const int m0 = blockIdx.y * BM, n0 = blockIdx.x * BN;
  const int wm = (w & 3) * 32, wn = (w >> 2) * 32;
  v8f acc[2][2] = {};
  const int nk = K / BK;

  for (int kb = 0; kb < nk; ++kb) {
    // stage A tile: 128x32 bf16 (2 x 16B per thread), B tile: 64x32 (1 x 16B)
#pragma unroll
    for (int c = 0; c < 2; ++c) {
      int chunk = t * 2 + c;                       // 0..511
      int row = chunk >> 2, col = (chunk & 3) * 8;
      const __bf16* src = &A[(size_t)(m0 + row) * K + kb * BK + col];
#if HAVE_ASYNC_LDS
      async_lds_b128(src, &As[row][col]);
#else
      *(v8bf*)&As[row][col] = *(const v8bf*)src;
#endif
    }
    {
      int row = t >> 2, col = (t & 3) * 8;
      const __bf16* src = &B[(size_t)(n0 + row) * K + kb * BK + col];
#if HAVE_ASYNC_LDS
      async_lds_b128(src, &Bs[row][col]);
#else
      *(v8bf*)&Bs[row][col] = *(const v8bf*)src;
#endif
    }
    if (kb + 1 < nk) {  // warm L2/GL1 for the next k-tile
      __builtin_prefetch(&A[(size_t)(m0 + (t >> 1)) * K + (kb + 1) * BK], 0, 0);
      __builtin_prefetch(&B[(size_t)(n0 + (t & 63)) * K + (kb + 1) * BK], 0, 0);
    }
#if HAVE_ASYNC_LDS
    wait_async();
#endif
    __syncthreads();

    v16bf af[2], bfv[2];
#pragma unroll
    for (int i = 0; i < 2; ++i) af[i] = load_frag(&As[wm + i * 16][0], LDT);
#pragma unroll
    for (int i = 0; i < 2; ++i) bfv[i] = load_frag(&Bs[wn + i * 16][0], LDT);
#pragma unroll
    for (int mi = 0; mi < 2; ++mi)
#pragma unroll
      for (int ni = 0; ni < 2; ++ni)
        acc[mi][ni] = __builtin_amdgcn_wmma_f32_16x16x32_bf16(
            false, af[mi], false, bfv[ni], (short)0, acc[mi][ni], false, false);
    __syncthreads();
  }

  // epilogue: C layout -> VGPR r = row (+8 for upper lane half), lane&15 = col
  const int half = lane >> 4, nl = lane & 15;
#pragma unroll
  for (int mi = 0; mi < 2; ++mi)
#pragma unroll
    for (int ni = 0; ni < 2; ++ni) {
      int col = n0 + wn + ni * 16 + nl;
      float bv = bias[col];
#pragma unroll
      for (int r = 0; r < 8; ++r) {
        int row = m0 + wm + mi * 16 + half * 8 + r;
        float v = acc[mi][ni][r] + bv;
        if (OUT_BF16)
          ((__bf16*)Cout)[(size_t)row * N + col] = f2bf(v);
        else
          ((float*)Cout)[(size_t)row * N + col] = v;
      }
    }
}

// ---------- kernel 2: flash attention over bf16 qkv ----------
// qkv: [8192, 1536]; per head h columns: q=h*192+0..63, k=+64, v=+128.
// Block = 8 waves x 16 query rows = 128 rows; grid = 32 (b,h) * 16 blocks.

__global__ __launch_bounds__(256) void flash_attn_wmma(
    const __bf16* __restrict__ qkv, __bf16* __restrict__ attn) {
  constexpr int LDT = 72;  // 144B rows: 16B aligned, bank-stagger
  __shared__ __align__(16) __bf16 Ks[64][LDT];        // K tile, row = key j
  __shared__ __align__(16) __bf16 Vts[64][LDT];       // V^T tile, row = d
  __shared__ __align__(16) __bf16 Ps[8][16][LDT];     // per-wave P (i x j)
  const int t = threadIdx.x, lane = t & 31, w = t >> 5;
  const int half = lane >> 4, nl = lane & 15;
  const int mblk = blockIdx.x & 15, bh = blockIdx.x >> 4;
  const int b = bh >> 3, h = bh & 7;
  const size_t rowbase = (size_t)b * SEQ;
  const int qcol = h * 3 * DK;
  const int m_w = mblk * 128 + w * 16;  // this wave's 16 query rows

  // Q fragments (persist across key loop): two k-halves d=0..31 / 32..63
  v16bf qa[2];
#pragma unroll
  for (int i = 0; i < 2; ++i)
    qa[i] = load_frag(qkv + (rowbase + m_w) * 1536 + qcol + i * 32, 1536);

  v8f acc[4] = {};       // O accumulators, d tiles 0,16,32,48
  float mrun[8], lrun[8];
#pragma unroll
  for (int r = 0; r < 8; ++r) { mrun[r] = -1e30f; lrun[r] = 0.f; }

  for (int kb = 0; kb < SEQ / 64; ++kb) {
    // cooperative stage: K rows async direct-to-LDS, V transposed via VGPRs
#pragma unroll
    for (int c = 0; c < 2; ++c) {
      int chunk = t * 2 + c;                   // 0..511
      int j = chunk >> 3, col = (chunk & 7) * 8;
      size_t src = (rowbase + kb * 64 + j) * 1536 + qcol;
#if HAVE_ASYNC_LDS
      async_lds_b128(&qkv[src + 64 + col], &Ks[j][col]);
#else
      *(v8bf*)&Ks[j][col] = *(const v8bf*)&qkv[src + 64 + col];
#endif
      v8bf vv = *(const v8bf*)&qkv[src + 128 + col];
#pragma unroll
      for (int e = 0; e < 8; ++e) Vts[col + e][j] = vv[e];
    }
#if HAVE_ASYNC_LDS
    wait_async();
#endif
    __syncthreads();

    // S = (Q K^T) * 1/sqrt(dk): 4 j-tiles x 2 k-steps
    v8f st[4] = {};
#pragma unroll
    for (int jt = 0; jt < 4; ++jt)
#pragma unroll
      for (int kh = 0; kh < 2; ++kh)
        st[jt] = __builtin_amdgcn_wmma_f32_16x16x32_bf16(
            false, qa[kh], false, load_frag(&Ks[jt * 16][kh * 32], LDT),
            (short)0, st[jt], false, false);

    // online softmax: row r lives in VGPR r across the 16-lane half
    float alpha[8];
#pragma unroll
    for (int r = 0; r < 8; ++r) {
      float tm = -1e30f;
#pragma unroll
      for (int jt = 0; jt < 4; ++jt) {
        st[jt][r] *= 0.125f;
        tm = fmaxf(tm, st[jt][r]);
      }
#pragma unroll
      for (int d = 1; d < 16; d <<= 1) tm = fmaxf(tm, __shfl_xor(tm, d, 32));
      float mn = fmaxf(mrun[r], tm);
      alpha[r] = __expf(mrun[r] - mn);
      mrun[r] = mn;
      float rs = 0.f;
#pragma unroll
      for (int jt = 0; jt < 4; ++jt) {
        float p = __expf(st[jt][r] - mn);
        st[jt][r] = p;
        rs += p;
        Ps[w][half * 8 + r][jt * 16 + nl] = f2bf(p);  // C-layout -> LDS
      }
#pragma unroll
      for (int d = 1; d < 16; d <<= 1) rs += __shfl_xor(rs, d, 32);
      lrun[r] = lrun[r] * alpha[r] + rs;
    }

    // rescale running O by alpha (per row)
#pragma unroll
    for (int dn = 0; dn < 4; ++dn)
#pragma unroll
      for (int r = 0; r < 8; ++r) acc[dn][r] *= alpha[r];

    // O += P @ V : A = P (k=j), B = V^T rows (lane = d, contiguous j)
    v16bf pa[2];
#pragma unroll
    for (int jh = 0; jh < 2; ++jh)
      pa[jh] = load_frag(&Ps[w][0][jh * 32], LDT);
#pragma unroll
    for (int dn = 0; dn < 4; ++dn)
#pragma unroll
      for (int jh = 0; jh < 2; ++jh)
        acc[dn] = __builtin_amdgcn_wmma_f32_16x16x32_bf16(
            false, pa[jh], false, load_frag(&Vts[dn * 16][jh * 32], LDT),
            (short)0, acc[dn], false, false);
    __syncthreads();
  }

  // normalize and store attn[token, h*64 + d] as bf16
#pragma unroll
  for (int dn = 0; dn < 4; ++dn)
#pragma unroll
    for (int r = 0; r < 8; ++r) {
      int row = m_w + half * 8 + r;
      attn[(rowbase + row) * D_MODEL + h * DK + dn * 16 + nl] =
          f2bf(acc[dn][r] / lrun[r]);
    }
}

// ---------- host launcher ----------

extern "C" void kernel_launch(void* const* d_in, const int* in_sizes, int n_in,
                              void* d_out, int out_size, void* d_ws,
                              size_t ws_size, hipStream_t stream) {
  (void)in_sizes; (void)n_in; (void)out_size; (void)ws_size;
  const float* x     = (const float*)d_in[0];
  const float* w_qkv = (const float*)d_in[1];
  const float* b_qkv = (const float*)d_in[2];
  const float* w_o   = (const float*)d_in[3];
  const float* b_o   = (const float*)d_in[4];

  char* ws = (char*)d_ws;
  size_t off = 0;
  auto take = [&](size_t bytes) -> char* {
    char* p = ws + off;
    off = (off + bytes + 255) & ~(size_t)255;
    return p;
  };
  __bf16* xb    = (__bf16*)take((size_t)MTOK * D_MODEL * 2);
  __bf16* wqb   = (__bf16*)take((size_t)3 * D_MODEL * D_MODEL * 2);
  __bf16* wob   = (__bf16*)take((size_t)D_MODEL * D_MODEL * 2);
  __bf16* qkvb  = (__bf16*)take((size_t)MTOK * 3 * D_MODEL * 2);
  __bf16* attnb = (__bf16*)take((size_t)MTOK * D_MODEL * 2);

  int nx = MTOK * D_MODEL;
  convert_f32_bf16<<<(nx + 255) / 256, 256, 0, stream>>>(x, xb, nx);
  int nwq = 3 * D_MODEL * D_MODEL;
  convert_f32_bf16<<<(nwq + 255) / 256, 256, 0, stream>>>(w_qkv, wqb, nwq);
  int nwo = D_MODEL * D_MODEL;
  convert_f32_bf16<<<(nwo + 255) / 256, 256, 0, stream>>>(w_o, wob, nwo);

  // QKV projection: [8192,512] x [1536,512]^T -> bf16 [8192,1536]
  gemm_bias_wmma<true><<<dim3(3 * D_MODEL / 64, MTOK / 128), 256, 0, stream>>>(
      xb, wqb, b_qkv, qkvb, MTOK, 3 * D_MODEL, D_MODEL);

  // flash attention: 32 (b,h) pairs x 16 row-blocks
  flash_attn_wmma<<<BATCH * HEADS * (SEQ / 128), 256, 0, stream>>>(qkvb, attnb);

  // output projection: [8192,512] x [512,512]^T + b_o -> f32 d_out
  gemm_bias_wmma<false><<<dim3(D_MODEL / 64, MTOK / 128), 256, 0, stream>>>(
      attnb, wob, b_o, d_out, MTOK, D_MODEL, D_MODEL);
}
